// SelfAttention_81166291960308
// MI455X (gfx1250) — matrix operand order repaired and checked by reading, not compiled, not run on hardware
//
#include <hip/hip_runtime.h>
#include <hip/hip_bf16.h>

// ---------------------------------------------------------------------------
// Types for CDNA5 WMMA (wave32): bf16 16x16x32, f32 accumulate.
// ---------------------------------------------------------------------------
typedef __attribute__((ext_vector_type(16))) __bf16 v16bf;
typedef __attribute__((ext_vector_type(8)))  __bf16 bf16x8;   // 16 bytes
typedef __attribute__((ext_vector_type(8)))  float  v8f;
typedef __attribute__((ext_vector_type(4)))  unsigned int v4u;
typedef __attribute__((ext_vector_type(8)))  int    v8i;
typedef __attribute__((ext_vector_type(4)))  int    v4i;

union FragU {
    v16bf  v;
    bf16x8 h[2];
};

__device__ __forceinline__ v8f wmma_bf16(v16bf a, v16bf b, v8f c) {
    // (neg_a, A, neg_b, B, c_mod, C, reuse_a, reuse_b)
    return __builtin_amdgcn_wmma_f32_16x16x32_bf16(false, a, false, b,
                                                   (short)0, c, false, false);
}

// ---------------------------------------------------------------------------
// Tensor Data Mover: DMA a 2D tile (tile_w x tile_h, 2-byte elements) from a
// row-major global matrix (row stride = row_stride_elems) into LDS at lds_off.
// Descriptor packing per CDNA5 ISA 8.3/8.4 (D# group0/group1), 2D tensor,
// groups 2/3 zero (VADDR2/3 conceptually NULL). 6-arg builtin form
// (amdgpu-toolchain clang-23): (g0, g1, g2, g3, g4, cpol).
// ---------------------------------------------------------------------------
__device__ __forceinline__ void tdm_load_2d(unsigned lds_off, const void* gaddr,
                                            unsigned tile_w_elems, unsigned tile_h,
                                            unsigned row_stride_elems) {
    unsigned long long ga = (unsigned long long)(size_t)gaddr;
    v4u g0;
    g0[0] = 1u;                                   // count=1, user descriptor
    g0[1] = lds_off;                              // lds_addr (bytes)
    g0[2] = (unsigned)(ga & 0xffffffffu);         // global_addr[31:0]
    g0[3] = (unsigned)((ga >> 32) & 0x01ffffffu)  // global_addr[56:32]
            | (2u << 30);                         // type = 2 ("image")
    const unsigned td0 = row_stride_elems;        // tensor dim0 (elements)
    const unsigned td1 = tile_h;                  // tensor dim1 (rows)
    const unsigned long long s0 = row_stride_elems; // dim0 stride (elements)
    v8i g1;
    g1[0] = 0x00010000;                           // wg_mask=0, data_size=1 (2B)
    g1[1] = (int)((td0 & 0xffffu) << 16);         // barrier_addr=0 | td0[15:0]
    g1[2] = (int)(((td0 >> 16) & 0xffffu) | ((td1 & 0xffffu) << 16));
    g1[3] = (int)(((td1 >> 16) & 0xffffu) | ((tile_w_elems & 0xffffu) << 16));
    g1[4] = (int)(tile_h & 0xffffu);              // tile_dim1 | tile_dim2=0
    g1[5] = (int)(s0 & 0xffffffffu);              // dim0_stride[31:0]
    g1[6] = (int)((s0 >> 32) & 0xffffu);          // dim0_stride[47:32] | d1s=0
    g1[7] = 0;
    const v4i z4 = {0, 0, 0, 0};
    const v8i z8 = {0, 0, 0, 0, 0, 0, 0, 0};
    __builtin_amdgcn_tensor_load_to_lds(g0, g1, z4, z4, z8, 0);
}

// ---------------------------------------------------------------------------
// Problem constants (4, 2048, 1024), 16 heads, d_head = 64.
// ---------------------------------------------------------------------------
#define BATCH   4
#define SEQ     2048
#define EDIM    1024
#define NHEAD   16
#define DHEAD   64
#define MTOT    (BATCH * SEQ)        // 8192
#define N1      (3 * EDIM)           // 3072

// ---------------------------------------------------------------------------
// fp32 -> bf16 conversion helpers
// ---------------------------------------------------------------------------
__global__ __launch_bounds__(256) void cvt_kernel(const float* __restrict__ in,
                                                  __bf16* __restrict__ out, int n) {
    int i = blockIdx.x * 256 + threadIdx.x;
    if (i < n) out[i] = (__bf16)in[i];
}

// out[(n,k)] = in[(k,n)] : store W transposed so B-fragments are row reads.
__global__ __launch_bounds__(256) void cvtT_kernel(const float* __restrict__ in,
                                                   __bf16* __restrict__ out,
                                                   int K, int N) {
    int i = blockIdx.x * 256 + threadIdx.x;
    if (i < K * N) {
        int n = i / K;
        int k = i - n * K;
        out[i] = (__bf16)in[(size_t)k * N + n];
    }
}

// ---------------------------------------------------------------------------
// Shared 128x128 GEMM mainloop, TDM double-buffered:
//   wave 0 DMAs the next 128x32 A/B tiles into the alternate LDS buffer while
//   all 8 waves run WMMAs on the current buffer. TDM completes in order, so
//   s_wait_tensorcnt(2) after issuing the next pair means the current pair
//   has landed. ldsA/ldsB each hold two 128x32 bf16 buffers (2 x 8KB).
// 8 waves: wave = (wm,wn), wm in 0..3 (rows of 32), wn in 0..1 (cols of 64).
// ---------------------------------------------------------------------------
__device__ __forceinline__ void gemm_tile(const __bf16* __restrict__ A,
                                          const __bf16* __restrict__ Bt,
                                          int K, int m0, int n0,
                                          __bf16* ldsA, __bf16* ldsB,
                                          v8f (&acc)[2][4]) {
    const int tid  = threadIdx.x;
    const int lane = tid & 31;
    const int wave = tid >> 5;
    const int wm = wave >> 1, wn = wave & 1;
    const int l15 = lane & 15, hi = lane >> 4;
    const unsigned ldsAoff = (unsigned)(size_t)ldsA;
    const unsigned ldsBoff = (unsigned)(size_t)ldsB;

    if (wave == 0) {     // prologue: DMA first tile pair into buffer 0
        tdm_load_2d(ldsAoff, A + (size_t)m0 * K, 32, 128, K);
        tdm_load_2d(ldsBoff, Bt + (size_t)n0 * K, 32, 128, K);
    }

    for (int k0 = 0; k0 < K; k0 += 32) {
        const int cur = (k0 >> 5) & 1;
        const int nxt = cur ^ 1;
        __syncthreads();                 // all waves done reading buffer `nxt`
        if (wave == 0) {
            if (k0 + 32 < K) {           // DMA next tile pair one step ahead
                tdm_load_2d(ldsAoff + (unsigned)nxt * 8192u,
                            A + (size_t)m0 * K + k0 + 32, 32, 128, K);
                tdm_load_2d(ldsBoff + (unsigned)nxt * 8192u,
                            Bt + (size_t)n0 * K + k0 + 32, 32, 128, K);
                __builtin_amdgcn_s_wait_tensorcnt(2);  // current pair landed
            } else {
                __builtin_amdgcn_s_wait_tensorcnt(0);
            }
        }
        __syncthreads();                 // current buffer visible to all waves

        const __bf16* curA = ldsA + cur * 4096;   // 4096 elems = 8KB
        const __bf16* curB = ldsB + cur * 4096;

        // A-fragments (16x32, ISA layout: half-wave 0 -> K 0-7 & 16-23).
        FragU af[2];
#pragma unroll
        for (int mt = 0; mt < 2; ++mt) {
            const __bf16* ap = &curA[(wm * 32 + mt * 16 + l15) * 32 + hi * 8];
            af[mt].h[0] = *(const bf16x8*)ap;
            af[mt].h[1] = *(const bf16x8*)(ap + 16);
        }
        // B-fragments (32x16): half-wave 0 -> K 0-7 & 16-23 of column l15.
        FragU bfr[4];
#pragma unroll
        for (int nt = 0; nt < 4; ++nt) {
            const __bf16* bp = &curB[(wn * 64 + nt * 16 + l15) * 32 + hi * 8];
            bfr[nt].h[0] = *(const bf16x8*)bp;
            bfr[nt].h[1] = *(const bf16x8*)(bp + 16);
        }
#pragma unroll
        for (int mt = 0; mt < 2; ++mt)
#pragma unroll
            for (int nt = 0; nt < 4; ++nt)
                acc[mt][nt] = wmma_bf16(af[mt].v, bfr[nt].v, acc[mt][nt]);
    }
}

// ---------------------------------------------------------------------------
// GEMM 1: h = x @ W1 + b1, scattered into Q (nh,s,64), K (nh,s,64), Vt (nh,64,s)
// ---------------------------------------------------------------------------
__global__ __launch_bounds__(256) void gemm_qkv_kernel(
        const __bf16* __restrict__ A, const __bf16* __restrict__ Bt,
        const float* __restrict__ b1,
        __bf16* __restrict__ Qb, __bf16* __restrict__ Kb, __bf16* __restrict__ Vt) {
    __shared__ alignas(128) __bf16 ldsA[2 * 128 * 32];
    __shared__ alignas(128) __bf16 ldsB[2 * 128 * 32];
    const int n0 = blockIdx.x * 128;
    const int m0 = blockIdx.y * 128;
    v8f acc[2][4] = {};
    gemm_tile(A, Bt, EDIM, m0, n0, ldsA, ldsB, acc);

    const int lane = threadIdx.x & 31, wave = threadIdx.x >> 5;
    const int wm = wave >> 1, wn = wave & 1, l15 = lane & 15, hi = lane >> 4;
#pragma unroll
    for (int mt = 0; mt < 2; ++mt) {
#pragma unroll
        for (int nt = 0; nt < 4; ++nt) {
            const int ncol = n0 + wn * 64 + nt * 16 + l15;
            const float bias = b1[ncol];
            const int head = ncol / 192;
            const int rr   = ncol - head * 192;
            const int sec  = rr >> 6;       // 0=q, 1=k, 2=v
            const int d    = rr & 63;
#pragma unroll
            for (int r = 0; r < 8; ++r) {
                const int m = m0 + wm * 32 + mt * 16 + r + 8 * hi;
                const int batch = m >> 11, s = m & (SEQ - 1);
                const int nh = batch * NHEAD + head;
                const __bf16 val = (__bf16)(acc[mt][nt][r] + bias);
                if (sec == 0)
                    Qb[((size_t)nh * SEQ + s) * DHEAD + d] = val;
                else if (sec == 1)
                    Kb[((size_t)nh * SEQ + s) * DHEAD + d] = val;
                else
                    Vt[((size_t)nh * DHEAD + d) * SEQ + s] = val;  // transposed
            }
        }
    }
}

// ---------------------------------------------------------------------------
// Flash attention: one wave owns 16 query rows; online softmax over k-chunks
// of 32. S via 4 WMMAs, P.V via 4 WMMAs per chunk. P goes through LDS to
// convert C-layout -> A-layout.
// ---------------------------------------------------------------------------
__global__ __launch_bounds__(256) void attn_kernel(
        const __bf16* __restrict__ Qb, const __bf16* __restrict__ Kb,
        const __bf16* __restrict__ Vt, __bf16* __restrict__ ctx) {
    __shared__ alignas(128) __bf16 Pbuf[8][16 * 32];
    const int nh   = blockIdx.x;            // 0..63 = batch*16 + head
    const int q0   = blockIdx.y * 128;      // query tile base
    const int lane = threadIdx.x & 31, wave = threadIdx.x >> 5;
    const int l15  = lane & 15, hi = lane >> 4;

    const __bf16* Qrow  = Qb + ((size_t)nh * SEQ + q0 + wave * 16 + l15) * DHEAD;
    const __bf16* Kbase = Kb + (size_t)nh * SEQ * DHEAD;
    const __bf16* Vbase = Vt + (size_t)nh * DHEAD * SEQ;
    __bf16* pl = &Pbuf[wave][0];

    // Persistent Q fragments: d 0..31 and d 32..63.
    FragU qf[2];
#pragma unroll
    for (int dt = 0; dt < 2; ++dt) {
        const int ko = dt * 32 + hi * 16;
        qf[dt].h[0] = *(const bf16x8*)(Qrow + ko);
        qf[dt].h[1] = *(const bf16x8*)(Qrow + ko + 8);
    }

    v8f   o[4] = {};
    float mrow[8], lsum[8];
#pragma unroll
    for (int r = 0; r < 8; ++r) { mrow[r] = -3.0e38f; lsum[r] = 0.0f; }

    for (int kb = 0; kb < SEQ; kb += 32) {
        // ---- S = Q @ K^T for 32 key columns (two 16-col WMMA tiles) ----
        v8f s0 = {}, s1 = {};
        {
            const __bf16* kr = Kbase + (size_t)(kb + l15) * DHEAD + hi * 16;
            FragU ka, kc;
            ka.h[0] = *(const bf16x8*)kr;        ka.h[1] = *(const bf16x8*)(kr + 8);
            kc.h[0] = *(const bf16x8*)(kr + 32); kc.h[1] = *(const bf16x8*)(kr + 40);
            s0 = wmma_bf16(qf[0].v, ka.v, s0);
            s0 = wmma_bf16(qf[1].v, kc.v, s0);
            if (kb + 32 < SEQ)                    // hint next K-chunk
                __builtin_prefetch(kr + 32 * DHEAD, 0, 1);
        }
        {
            const __bf16* kr = Kbase + (size_t)(kb + 16 + l15) * DHEAD + hi * 16;
            FragU ka, kc;
            ka.h[0] = *(const bf16x8*)kr;        ka.h[1] = *(const bf16x8*)(kr + 8);
            kc.h[0] = *(const bf16x8*)(kr + 32); kc.h[1] = *(const bf16x8*)(kr + 40);
            s1 = wmma_bf16(qf[0].v, ka.v, s1);
            s1 = wmma_bf16(qf[1].v, kc.v, s1);
        }
        // ---- online softmax per query row (row = r + 8*hi) ----
#pragma unroll
        for (int r = 0; r < 8; ++r) {
            const float sa = s0[r] * 0.125f;       // 1/sqrt(64)
            const float sb = s1[r] * 0.125f;
            float cm = fmaxf(sa, sb);
#pragma unroll
            for (int off = 1; off < 16; off <<= 1)
                cm = fmaxf(cm, __shfl_xor(cm, off, 32));
            const float mn   = fmaxf(mrow[r], cm);
            const float corr = __expf(mrow[r] - mn);
            const float pa   = __expf(sa - mn);
            const float pbv  = __expf(sb - mn);
            float rs = pa + pbv;
#pragma unroll
            for (int off = 1; off < 16; off <<= 1)
                rs += __shfl_xor(rs, off, 32);
            lsum[r] = lsum[r] * corr + rs;
            mrow[r] = mn;
#pragma unroll
            for (int dt = 0; dt < 4; ++dt)
                o[dt][r] = o[dt][r] * corr;
            const int prow = r + 8 * hi;
            pl[prow * 32 + l15]      = (__bf16)pa;
            pl[prow * 32 + 16 + l15] = (__bf16)pbv;
        }
        // ---- P (C-layout in LDS) -> A-fragment, then O += P @ V ----
        FragU pf;
        pf.h[0] = *(const bf16x8*)(pl + l15 * 32 + hi * 16);
        pf.h[1] = *(const bf16x8*)(pl + l15 * 32 + hi * 16 + 8);
#pragma unroll
        for (int dt = 0; dt < 4; ++dt) {
            const __bf16* vr = Vbase + (size_t)(dt * 16 + l15) * SEQ + kb + hi * 16;
            FragU vf;
            vf.h[0] = *(const bf16x8*)vr;
            vf.h[1] = *(const bf16x8*)(vr + 8);
            o[dt] = wmma_bf16(pf.v, vf.v, o[dt]);
            if (kb + 32 < SEQ)
                __builtin_prefetch(vr + 32, 0, 1);
        }
    }

    // ---- epilogue: normalize and write ctx in (batch, s, E) layout ----
    const int batch = nh >> 4, head = nh & 15;
#pragma unroll
    for (int dt = 0; dt < 4; ++dt) {
#pragma unroll
        for (int r = 0; r < 8; ++r) {
            const int srow = q0 + wave * 16 + r + 8 * hi;
            const float val = o[dt][r] / lsum[r];
            ctx[((size_t)batch * SEQ + srow) * EDIM + head * DHEAD + dt * 16 + l15] =
                (__bf16)val;
        }
    }
}

// ---------------------------------------------------------------------------
// GEMM 2: out = ctx @ W2 + b2 (fp32 output)
// ---------------------------------------------------------------------------
__global__ __launch_bounds__(256) void gemm_out_kernel(
        const __bf16* __restrict__ A, const __bf16* __restrict__ Bt,
        const float* __restrict__ b2, float* __restrict__ out) {
    __shared__ alignas(128) __bf16 ldsA[2 * 128 * 32];
    __shared__ alignas(128) __bf16 ldsB[2 * 128 * 32];
    const int n0 = blockIdx.x * 128;
    const int m0 = blockIdx.y * 128;
    v8f acc[2][4] = {};
    gemm_tile(A, Bt, EDIM, m0, n0, ldsA, ldsB, acc);

    const int lane = threadIdx.x & 31, wave = threadIdx.x >> 5;
    const int wm = wave >> 1, wn = wave & 1, l15 = lane & 15, hi = lane >> 4;
#pragma unroll
    for (int mt = 0; mt < 2; ++mt) {
#pragma unroll
        for (int nt = 0; nt < 4; ++nt) {
            const int ncol = n0 + wn * 64 + nt * 16 + l15;
            const float bias = b2[ncol];
#pragma unroll
            for (int r = 0; r < 8; ++r) {
                const int m = m0 + wm * 32 + mt * 16 + r + 8 * hi;
                out[(size_t)m * EDIM + ncol] = acc[mt][nt][r] + bias;
            }
        }
    }
}

// ---------------------------------------------------------------------------
// Launcher
// ---------------------------------------------------------------------------
extern "C" void kernel_launch(void* const* d_in, const int* in_sizes, int n_in,
                              void* d_out, int out_size, void* d_ws, size_t ws_size,
                              hipStream_t stream) {
    (void)in_sizes; (void)n_in; (void)out_size; (void)ws_size;
    const float* x  = (const float*)d_in[0];   // (4,2048,1024)
    const float* W1 = (const float*)d_in[1];   // (1024,3072)
    const float* b1 = (const float*)d_in[2];   // (3072,)
    const float* W2 = (const float*)d_in[3];   // (1024,1024)
    const float* b2 = (const float*)d_in[4];   // (1024,)
    float* out = (float*)d_out;                // (4,2048,1024)

    char* ws = (char*)d_ws;
    __bf16* xb   = (__bf16*)(ws);                         // 16 MB
    __bf16* W1bT = (__bf16*)(ws + 16777216);              //  6 MB (3072x1024)
    __bf16* W2bT = (__bf16*)(ws + 23068672);              //  2 MB (1024x1024)
    __bf16* Qb   = (__bf16*)(ws + 25165824);              // 16 MB (64,2048,64)
    __bf16* Kb   = (__bf16*)(ws + 41943040);              // 16 MB (64,2048,64)
    __bf16* Vt   = (__bf16*)(ws + 58720256);              // 16 MB (64,64,2048)
    __bf16* ctx  = (__bf16*)(ws + 75497472);              // 16 MB (8192,1024)

    // 1) precision conversion (+ weight transposes)
    cvt_kernel <<<(MTOT * EDIM + 255) / 256, 256, 0, stream>>>(x, xb, MTOT * EDIM);
    cvtT_kernel<<<(EDIM * N1   + 255) / 256, 256, 0, stream>>>(W1, W1bT, EDIM, N1);
    cvtT_kernel<<<(EDIM * EDIM + 255) / 256, 256, 0, stream>>>(W2, W2bT, EDIM, EDIM);

    // 2) QKV projection, scatter per head
    gemm_qkv_kernel<<<dim3(N1 / 128, MTOT / 128), 256, 0, stream>>>(
        xb, W1bT, b1, Qb, Kb, Vt);

    // 3) flash attention per (batch*head, q-tile)
    attn_kernel<<<dim3(BATCH * NHEAD, SEQ / 128), 256, 0, stream>>>(Qb, Kb, Vt, ctx);

    // 4) output projection
    gemm_out_kernel<<<dim3(EDIM / 128, MTOT / 128), 256, 0, stream>>>(
        ctx, W2bT, b2, out);
}
